// pureGIN_88364657148568
// MI455X (gfx1250) — matrix operand order repaired and checked by reading, not compiled, not run on hardware
//
#include <hip/hip_runtime.h>
#include <hip/hip_bf16.h>

typedef __bf16 bf16;
typedef __attribute__((ext_vector_type(16))) __bf16 v16bf;
typedef __attribute__((ext_vector_type(8)))  __bf16 v8bf;
typedef __attribute__((ext_vector_type(8)))  float  v8f;

// ---------------------------------------------------------------- utilities
__global__ __launch_bounds__(256) void k_copy_f4(const float4* __restrict__ in,
                                                 float4* __restrict__ out, int n4) {
  int i = blockIdx.x * 256 + threadIdx.x;
  if (i < n4) out[i] = in[i];
}

__global__ __launch_bounds__(256) void k_zero_f(float* __restrict__ p, int n) {
  int i = blockIdx.x * 256 + threadIdx.x;
  if (i < n) p[i] = 0.0f;
}

// ------------------------------------------------- edge scatter-add (L2 atomics)
// H[dst[e]][:] += Xsrc[src[e]][:], C/4 threads per edge, float4 per thread.
__global__ __launch_bounds__(256) void k_edge_aggr(const int* __restrict__ src,
                                                   const int* __restrict__ dst,
                                                   const float* __restrict__ Xsrc,
                                                   float* __restrict__ H,
                                                   int nEdges, int C) {
  int tpe = C >> 2;  // threads per edge
  long long gid = (long long)blockIdx.x * 256 + threadIdx.x;
  long long total = (long long)nEdges * tpe;
  if (gid >= total) return;
  int e = (int)(gid / tpe);
  int q = (int)(gid - (long long)e * tpe);
  int s = src[e];
  int d = dst[e];
  float4 v = *(const float4*)(Xsrc + (size_t)s * C + q * 4);
  float* hp = H + (size_t)d * C + q * 4;
  unsafeAtomicAdd(hp + 0, v.x);
  unsafeAtomicAdd(hp + 1, v.y);
  unsafeAtomicAdd(hp + 2, v.z);
  unsafeAtomicAdd(hp + 3, v.w);
}

// ------------------------------------------------------- WMMA helpers
// A fragment (16x32 bf16): per lane, K runs [kb0,kb0+15].
__device__ __forceinline__ v16bf load_a_from_f32(const float* __restrict__ rp, int kb0) {
  float4 f0 = *(const float4*)(rp + kb0);
  float4 f1 = *(const float4*)(rp + kb0 + 4);
  float4 f2 = *(const float4*)(rp + kb0 + 8);
  float4 f3 = *(const float4*)(rp + kb0 + 12);
  v16bf a;
  a[0]  = (bf16)f0.x; a[1]  = (bf16)f0.y; a[2]  = (bf16)f0.z; a[3]  = (bf16)f0.w;
  a[4]  = (bf16)f1.x; a[5]  = (bf16)f1.y; a[6]  = (bf16)f1.z; a[7]  = (bf16)f1.w;
  a[8]  = (bf16)f2.x; a[9]  = (bf16)f2.y; a[10] = (bf16)f2.z; a[11] = (bf16)f2.w;
  a[12] = (bf16)f3.x; a[13] = (bf16)f3.y; a[14] = (bf16)f3.z; a[15] = (bf16)f3.w;
  return a;
}

// ------------------------------------------------- fused 2-layer GIN MLP (WMMA)
// Xout[N,64] = relu( relu(H[N,Cin] @ W1 + b1) @ W2 + b2 )
// one wave per 16-row tile; weights bf16-transposed in LDS; T staged per-wave in LDS.
__global__ __launch_bounds__(256) void k_gin_mlp(const float* __restrict__ H,
                                                 const float* __restrict__ W1,
                                                 const float* __restrict__ B1,
                                                 const float* __restrict__ W2,
                                                 const float* __restrict__ B2,
                                                 float* __restrict__ Xout,
                                                 int N, int Cin) {
  __shared__ __align__(32) bf16  sW1t[64 * 128];   // [n][k], k stride = Cin
  __shared__ __align__(32) bf16  sW2t[64 * 64];    // [n][k]
  __shared__ __align__(32) bf16  sT[8 * 16 * 64];  // per-wave 16x64 intermediate
  __shared__ float sB1[64], sB2[64];

  const int tid = threadIdx.x;
  for (int i = tid; i < Cin * 64; i += 256) {      // W1 f32 -> bf16, transpose
    int k = i >> 6, n = i & 63;
    sW1t[n * Cin + k] = (bf16)W1[i];
  }
  for (int i = tid; i < 64 * 64; i += 256) {       // W2 f32 -> bf16, transpose
    int k = i >> 6, n = i & 63;
    sW2t[(n << 6) + k] = (bf16)W2[i];
  }
  if (tid < 64) { sB1[tid] = B1[tid]; sB2[tid] = B2[tid]; }
  __syncthreads();

  const int wave = tid >> 5;
  const int lane = tid & 31;
  const int half = lane >> 4;     // 0/1
  const int l16  = lane & 15;
  bf16* sTw = sT + wave * (16 * 64);

  const int m0 = (blockIdx.x * 8 + wave) * 16;
  if (m0 >= N) return;            // wave-uniform exit (EXEC stays all-ones for WMMA)

  const int arow = (m0 + l16 < N) ? (m0 + l16) : (N - 1);
  const float* rp = H + (size_t)arow * Cin;

  // ---- stage 1: T = relu(h @ W1 + b1), accumulated in 4 N-tiles of 16 cols
  v8f acc[4] = {{}, {}, {}, {}};
  for (int kk = 0; kk < Cin; kk += 32) {
    v16bf aF = load_a_from_f32(rp, kk + 16 * half);
#pragma unroll
    for (int nt = 0; nt < 4; ++nt) {
      int ncol = nt * 16 + l16;
      v16bf bF = *(const v16bf*)(sW1t + ncol * Cin + kk + 16 * half);
      acc[nt] = __builtin_amdgcn_wmma_f32_16x16x32_bf16(
          false, aF, false, bF, (short)0, acc[nt], false, false);
    }
  }
#pragma unroll
  for (int nt = 0; nt < 4; ++nt) {
    int n = nt * 16 + l16;
    float b = sB1[n];
#pragma unroll
    for (int r = 0; r < 8; ++r) {
      int m = r + 8 * half;
      float t = acc[nt][r] + b;
      sTw[m * 64 + n] = (bf16)fmaxf(t, 0.0f);   // same-wave DS: in-order, no barrier
    }
  }

  // ---- stage 2: U = relu(T @ W2 + b2)
  v8f acc2[4] = {{}, {}, {}, {}};
  for (int kk = 0; kk < 64; kk += 32) {
    int kb0 = kk + 16 * half;
    v8bf lo = *(const v8bf*)(sTw + l16 * 64 + kb0);
    v8bf hi = *(const v8bf*)(sTw + l16 * 64 + kb0 + 8);
    v16bf aF;
#pragma unroll
    for (int j = 0; j < 8; ++j) { aF[j] = lo[j]; aF[8 + j] = hi[j]; }
#pragma unroll
    for (int nt = 0; nt < 4; ++nt) {
      int ncol = nt * 16 + l16;
      v16bf bF = *(const v16bf*)(sW2t + (ncol << 6) + kk + 16 * half);
      acc2[nt] = __builtin_amdgcn_wmma_f32_16x16x32_bf16(
          false, aF, false, bF, (short)0, acc2[nt], false, false);
    }
  }
#pragma unroll
  for (int nt = 0; nt < 4; ++nt) {
    int n = nt * 16 + l16;
    float b = sB2[n];
#pragma unroll
    for (int r = 0; r < 8; ++r) {
      int row = m0 + r + 8 * half;
      if (row < N) Xout[(size_t)row * 64 + n] = fmaxf(acc2[nt][r] + b, 0.0f);
    }
  }
}

// ------------------------------------------------- global_add_pool via atomics
__global__ __launch_bounds__(256) void k_pool(const float* __restrict__ X,
                                              const int* __restrict__ batch,
                                              float* __restrict__ G, int N) {
  long long gid = (long long)blockIdx.x * 256 + threadIdx.x;
  if (gid >= (long long)N * 16) return;
  int node = (int)(gid >> 4);
  int q = (int)(gid & 15);
  int b = batch[node];
  float4 v = *(const float4*)(X + (size_t)node * 64 + q * 4);
  float* gp = G + (size_t)b * 64 + q * 4;
  unsafeAtomicAdd(gp + 0, v.x);
  unsafeAtomicAdd(gp + 1, v.y);
  unsafeAtomicAdd(gp + 2, v.z);
  unsafeAtomicAdd(gp + 3, v.w);
}

// ------------------------------------------------- tiny readout head
__global__ __launch_bounds__(64) void k_final(const float* __restrict__ G,
                                              const float* __restrict__ W1,
                                              const float* __restrict__ B1,
                                              const float* __restrict__ W2,
                                              const float* __restrict__ B2,
                                              float* __restrict__ out, int OUT) {
  __shared__ float hid[64];
  int g = blockIdx.x, t = threadIdx.x;
  const float* gr = G + (size_t)g * 64;
  float acc = B1[t];
  for (int k = 0; k < 64; ++k) acc += gr[k] * W1[k * 64 + t];
  hid[t] = fmaxf(acc, 0.0f);
  __syncthreads();
  if (t < OUT) {
    float o = B2[t];
    for (int k = 0; k < 64; ++k) o += hid[k] * W2[k * OUT + t];
    out[(size_t)g * OUT + t] = o;
  }
}

// ---------------------------------------------------------------- launcher
extern "C" void kernel_launch(void* const* d_in, const int* in_sizes, int n_in,
                              void* d_out, int out_size, void* d_ws, size_t ws_size,
                              hipStream_t stream) {
  const float* x      = (const float*)d_in[0];
  const int*   ei     = (const int*)d_in[1];
  const int*   batch  = (const int*)d_in[2];
  const float* w1_0   = (const float*)d_in[4];
  const float* b1_0   = (const float*)d_in[5];
  const float* w2_0   = (const float*)d_in[6];
  const float* b2_0   = (const float*)d_in[7];
  const float* ws1    = (const float*)d_in[8];
  const float* bs1    = (const float*)d_in[9];
  const float* ws2    = (const float*)d_in[10];
  const float* bs2    = (const float*)d_in[11];
  const float* mlp_w1 = (const float*)d_in[12];
  const float* mlp_b1 = (const float*)d_in[13];
  const float* mlp_w2 = (const float*)d_in[14];
  const float* mlp_b2 = (const float*)d_in[15];

  const int N    = in_sizes[2];            // 100000
  const int E    = in_sizes[1] / 2;        // 3200000
  const int Cin0 = in_sizes[0] / N;        // 128
  const int OUT  = in_sizes[15];           // 10
  const int NG   = out_size / OUT;         // 128
  const int*   src = ei;
  const int*   dst = ei + E;

  size_t hBytes = ((size_t)N * Cin0 * sizeof(float) + 255) & ~(size_t)255;
  size_t xBytes = ((size_t)N * 64 * sizeof(float) + 255) & ~(size_t)255;
  float* H = (float*)d_ws;
  float* X = (float*)((char*)d_ws + hBytes);
  float* G = (float*)((char*)d_ws + hBytes + xBytes);

  const int tiles    = (N + 15) / 16;
  const int mlpBlks  = (tiles + 7) / 8;

  // ---- layer 0 (Cin = 128)
  {
    int n4 = N * Cin0 / 4;
    k_copy_f4<<<(n4 + 255) / 256, 256, 0, stream>>>((const float4*)x, (float4*)H, n4);
    long long tw = (long long)E * (Cin0 / 4);
    k_edge_aggr<<<(unsigned)((tw + 255) / 256), 256, 0, stream>>>(src, dst, x, H, E, Cin0);
    k_gin_mlp<<<mlpBlks, 256, 0, stream>>>(H, w1_0, b1_0, w2_0, b2_0, X, N, Cin0);
  }

  // ---- layers 1..4 (Cin = 64)
  for (int i = 0; i < 4; ++i) {
    int n4 = N * 64 / 4;
    k_copy_f4<<<(n4 + 255) / 256, 256, 0, stream>>>((const float4*)X, (float4*)H, n4);
    long long tw = (long long)E * 16;
    k_edge_aggr<<<(unsigned)((tw + 255) / 256), 256, 0, stream>>>(src, dst, X, H, E, 64);
    k_gin_mlp<<<mlpBlks, 256, 0, stream>>>(H, ws1 + i * 64 * 64, bs1 + i * 64,
                                           ws2 + i * 64 * 64, bs2 + i * 64, X, N, 64);
  }

  // ---- pooling + head
  int gElems = NG * 64;
  k_zero_f<<<(gElems + 255) / 256, 256, 0, stream>>>(G, gElems);
  long long pw = (long long)N * 16;
  k_pool<<<(unsigned)((pw + 255) / 256), 256, 0, stream>>>(X, batch, G, N);
  k_final<<<NG, 64, 0, stream>>>(G, mlp_w1, mlp_b1, mlp_w2, mlp_b2, (float*)d_out, OUT);
}
